// AttentionBlock_22162031247785
// MI455X (gfx1250) — matrix operand order repaired and checked by reading, not compiled, not run on hardware
//
#include <hip/hip_runtime.h>
#include <hip/hip_bf16.h>

// ---------------------------------------------------------------------------
// AttentionBlock for MI455X (gfx1250): bf16 WMMA + async global->LDS copies.
//   B=8, S=2048, D_MODEL=1024, D_K=D_V=64
// Stage 1: qp/kp/vp = proj(x)      (WMMA GEMM, fp32 -> bf16 workspace)
// Stage 2: flash attention          (WMMA QK^T + online softmax + WMMA PV,
//                                    V tiles via double-buffered async->LDS)
// Stage 3: out = o @ Wo + bo        (WMMA GEMM, fp32 output)
// ---------------------------------------------------------------------------

typedef __attribute__((ext_vector_type(16))) __bf16 v16bf;
typedef __attribute__((ext_vector_type(8)))  float  v8f;

#define WMMA_BF16(a, b, c) \
  __builtin_amdgcn_wmma_f32_16x16x32_bf16(false, (a), false, (b), (short)0, (c), false, false)

__device__ __forceinline__ __bf16 f2bf(float f) { return (__bf16)f; }

// Async 16-byte global -> LDS copy (ASYNCcnt-tracked). LDS offset is the
// truncated flat address.
__device__ __forceinline__ void async_g2l_b128(void* lds, const void* g) {
  unsigned           l  = (unsigned)(uintptr_t)lds;
  unsigned long long ga = (unsigned long long)(uintptr_t)g;
  asm volatile("global_load_async_to_lds_b128 %0, %1, off"
               :: "v"(l), "v"(ga) : "memory");
}
__device__ __forceinline__ void async_wait1() {
  asm volatile("s_wait_asynccnt 0x1" ::: "memory");
}
__device__ __forceinline__ void async_wait0() {
  asm volatile("s_wait_asynccnt 0x0" ::: "memory");
}
__device__ __forceinline__ void ds_fence() {       // wave-local LDS ordering
  asm volatile("s_wait_dscnt 0x0" ::: "memory");
}

// ---------------------------------------------------------------------------
// Stage 1: Y[M,64](bf16) = X[M,K](f32) @ W[K,64](f32) + bias
// block = 128 threads = 4 waves; wave = 16 rows x 64 cols; grid.x = M/64
// ---------------------------------------------------------------------------
__global__ void proj_gemm(const float* __restrict__ X, const float* __restrict__ W,
                          const float* __restrict__ bias, __bf16* __restrict__ Y,
                          int K) {
  constexpr int N = 64;
  const int wave = threadIdx.x >> 5;
  const int lane = threadIdx.x & 31;
  const int hi   = lane >> 4;
  const int ln   = lane & 15;
  const int rowBase = blockIdx.x * 64 + wave * 16;
  const int row = rowBase + ln;

  v8f c0 = {}, c1 = {}, c2 = {}, c3 = {};

  for (int k0 = 0; k0 < K; k0 += 32) {
    // A fragment: row per lane, K interleave {c..c+7, c+16..c+23}, c = hi*8
    v16bf a;
    {
      const float* xp = X + (size_t)row * K + k0 + hi * 8;
#pragma unroll
      for (int i = 0; i < 8; ++i) { a[i] = f2bf(xp[i]); a[8 + i] = f2bf(xp[16 + i]); }
    }
    // B fragments: lane holds column ln (+16*nb), K interleave {c..c+7, c+16..c+23}, c = hi*8
    const float* wp = W + (size_t)(k0 + hi * 8) * N + ln;
    v16bf b0, b1, b2, b3;
#pragma unroll
    for (int i = 0; i < 8; ++i) {
      b0[i] = f2bf(wp[i * N]);      b1[i] = f2bf(wp[i * N + 16]);
      b2[i] = f2bf(wp[i * N + 32]); b3[i] = f2bf(wp[i * N + 48]);
      b0[8 + i] = f2bf(wp[(16 + i) * N]);      b1[8 + i] = f2bf(wp[(16 + i) * N + 16]);
      b2[8 + i] = f2bf(wp[(16 + i) * N + 32]); b3[8 + i] = f2bf(wp[(16 + i) * N + 48]);
    }
    c0 = WMMA_BF16(a, b0, c0);
    c1 = WMMA_BF16(a, b1, c1);
    c2 = WMMA_BF16(a, b2, c2);
    c3 = WMMA_BF16(a, b3, c3);
  }

  // C layout: VGPR r -> row rowBase + r + 8*hi, col = ln (+16*nb)
#pragma unroll
  for (int r = 0; r < 8; ++r) {
    const int m = rowBase + r + 8 * hi;
    __bf16* yp = Y + (size_t)m * N;
    yp[ln]      = f2bf(c0[r] + bias[ln]);
    yp[16 + ln] = f2bf(c1[r] + bias[16 + ln]);
    yp[32 + ln] = f2bf(c2[r] + bias[32 + ln]);
    yp[48 + ln] = f2bf(c3[r] + bias[48 + ln]);
  }
}

// ---------------------------------------------------------------------------
// Stage 2: flash attention, head dim 64, key tiles of 32.
// block = 128 threads = 4 waves; wave = 16 q rows; grid = (S/64, B)
// ---------------------------------------------------------------------------
__global__ void flash_attn(const __bf16* __restrict__ qp, const __bf16* __restrict__ kp,
                           const __bf16* __restrict__ vp, const int* __restrict__ mask,
                           __bf16* __restrict__ O, int S) {
  constexpr int D    = 64;
  constexpr int KT   = 32;   // key-tile size
  constexpr int KSTR = 36;   // padded row stride (bf16): 18 dwords -> no bank alias in wave
  constexpr int VSTR = 72;   // padded V row stride: 144B (16B-aligned chunks), 2-way alias only

  __shared__ __bf16 ldsKT[D * KSTR];         // K^T tile [d][key]
  __shared__ __bf16 ldsV[2][KT * VSTR];      // double-buffered V tile [key][d]
  __shared__ __bf16 ldsP[4][16 * KSTR];      // per-wave P staging

  const int wave = threadIdx.x >> 5;
  const int lane = threadIdx.x & 31;
  const int hi   = lane >> 4;
  const int ln   = lane & 15;
  const int b    = blockIdx.y;
  const int qBase = blockIdx.x * 64 + wave * 16;

  const __bf16* qb = qp + ((size_t)b * S + qBase) * D;
  const __bf16* kb = kp + (size_t)b * S * D;
  const __bf16* vb = vp + (size_t)b * S * D;

  // Issue the async copy of one 32x64 V tile into ldsV[buf] (2 chunks/thread).
  auto issueV = [&](int kt, int buf) {
#pragma unroll
    for (int c = 0; c < 2; ++c) {
      const int chunk = (int)threadIdx.x + c * 128;   // 256 chunks of 16B
      const int key = chunk >> 3, seg = chunk & 7;
      async_g2l_b128(&ldsV[buf][key * VSTR + seg * 8],
                     vb + (size_t)(kt + key) * D + seg * 8);
    }
  };

  // Q A-fragments: aQ0 covers d=0..31, aQ1 covers d=32..63
  v16bf aQ0, aQ1;
  {
    const __bf16* qrow = qb + (size_t)ln * D;
    const int c = hi * 8;
#pragma unroll
    for (int i = 0; i < 8; ++i) {
      aQ0[i]     = qrow[c + i];       aQ0[8 + i] = qrow[c + 16 + i];
      aQ1[i]     = qrow[32 + c + i];  aQ1[8 + i] = qrow[32 + c + 16 + i];
    }
  }

  v8f oA[4] = {};
  float m_r[8], l_r[8];
#pragma unroll
  for (int r = 0; r < 8; ++r) { m_r[r] = -3.0e38f; l_r[r] = 0.f; }

  issueV(0, 0);                      // prime the V pipeline

  for (int kt = 0; kt < S; kt += KT) {
    const int buf = (kt / KT) & 1;
    __syncthreads();                 // prior-iteration LDS reads are done

    // ---- stage K^T tile into LDS: ldsKT[d][key] (transposing store) ----
    {
      const int key  = threadIdx.x & 31;
      const int dblk = threadIdx.x >> 5;
      const __bf16* kr = kb + (size_t)(kt + key) * D + dblk * 16;
#pragma unroll
      for (int i = 0; i < 16; ++i)
        ldsKT[(dblk * 16 + i) * KSTR + key] = kr[i];
    }
    if (kt + KT < S) {
      __builtin_prefetch(kb + (size_t)(kt + KT + lane) * D, 0, 1);
      issueV(kt + KT, buf ^ 1);      // overlap next V copy with this tile's math
      async_wait1();                 // in-order completion => tile kt has landed
    } else {
      async_wait0();
    }
    __syncthreads();                 // K^T + V staging visible block-wide

    // ---- scores: two 16-key subtiles, contraction over d=64 ----
    v8f s0 = {}, s1 = {};
#pragma unroll
    for (int dstep = 0; dstep < 2; ++dstep) {
      v16bf bK0, bK1;
      const __bf16* kl = &ldsKT[(dstep * 32 + hi * 8) * KSTR + ln];
#pragma unroll
      for (int i = 0; i < 8; ++i) {
        bK0[i] = kl[i * KSTR];            bK1[i] = kl[i * KSTR + 16];
        bK0[8 + i] = kl[(16 + i) * KSTR]; bK1[8 + i] = kl[(16 + i) * KSTR + 16];
      }
      s0 = WMMA_BF16(dstep ? aQ1 : aQ0, bK0, s0);
      s1 = WMMA_BF16(dstep ? aQ1 : aQ0, bK1, s1);
    }

    // ---- mask + online softmax (row = r + 8*hi) ----
    float p0[8], p1[8];
#pragma unroll
    for (int r = 0; r < 8; ++r) {
      const int qrow = qBase + r + 8 * hi;
      const size_t mrow = (size_t)qrow * S + kt;
      float x0 = mask[mrow + ln]      ? s0[r] : -1.0e9f;
      float x1 = mask[mrow + 16 + ln] ? s1[r] : -1.0e9f;
      float mx = fmaxf(x0, x1);
#pragma unroll
      for (int off = 8; off; off >>= 1) mx = fmaxf(mx, __shfl_xor(mx, off, 32));
      const float mnew  = fmaxf(m_r[r], mx);
      const float scale = __expf(m_r[r] - mnew);
      const float e0 = __expf(x0 - mnew), e1 = __expf(x1 - mnew);
      float rsum = e0 + e1;
#pragma unroll
      for (int off = 8; off; off >>= 1) rsum += __shfl_xor(rsum, off, 32);
      l_r[r] = l_r[r] * scale + rsum;
      m_r[r] = mnew;
#pragma unroll
      for (int nb = 0; nb < 4; ++nb) oA[nb][r] *= scale;
      p0[r] = e0; p1[r] = e1;
    }

    // ---- transpose P through wave-private LDS (same-wave DS ops are in-order)
    __bf16* pw = &ldsP[wave][0];
#pragma unroll
    for (int r = 0; r < 8; ++r) {
      pw[(r + 8 * hi) * KSTR + ln]      = f2bf(p0[r]);
      pw[(r + 8 * hi) * KSTR + 16 + ln] = f2bf(p1[r]);
    }
    ds_fence();

    v16bf aP;
    {
      const __bf16* pr = pw + ln * KSTR + hi * 8;
#pragma unroll
      for (int i = 0; i < 8; ++i) { aP[i] = pr[i]; aP[8 + i] = pr[16 + i]; }
    }

    // ---- P @ V from LDS: lane = d column, keys interleaved {c..c+7, c+16..c+23}, c = hi*8 ----
    const __bf16* vrow = &ldsV[buf][hi * 8 * VSTR + ln];
#pragma unroll
    for (int nb = 0; nb < 4; ++nb) {
      v16bf bV;
#pragma unroll
      for (int i = 0; i < 8; ++i) { bV[i] = vrow[i * VSTR + nb * 16]; bV[8 + i] = vrow[(16 + i) * VSTR + nb * 16]; }
      oA[nb] = WMMA_BF16(aP, bV, oA[nb]);
    }
  }

  // ---- normalize and write O (bf16) ----
#pragma unroll
  for (int r = 0; r < 8; ++r) {
    const float inv = 1.0f / l_r[r];
    __bf16* op = O + ((size_t)b * S + qBase + r + 8 * hi) * D;
#pragma unroll
    for (int nb = 0; nb < 4; ++nb)
      op[nb * 16 + ln] = f2bf(oA[nb][r] * inv);
  }
}

// ---------------------------------------------------------------------------
// Stage 3: Y[M,1024](f32) = O[M,64](bf16) @ Wo[64,1024](f32) + bo
// block = 128 threads = 4 waves; block tile = 16 rows x 256 cols
// ---------------------------------------------------------------------------
__global__ void out_proj(const __bf16* __restrict__ Oin, const float* __restrict__ Wo,
                         const float* __restrict__ bo, float* __restrict__ Y, int N) {
  const int wave = threadIdx.x >> 5;
  const int lane = threadIdx.x & 31;
  const int hi   = lane >> 4;
  const int ln   = lane & 15;
  const int rowBase = blockIdx.x * 16;
  const int nBase   = blockIdx.y * 256 + wave * 64;

  v16bf a0, a1;
  {
    const __bf16* orow = Oin + (size_t)(rowBase + ln) * 64;
    const int c = hi * 8;
#pragma unroll
    for (int i = 0; i < 8; ++i) {
      a0[i]     = orow[c + i];      a0[8 + i] = orow[c + 16 + i];
      a1[i]     = orow[32 + c + i]; a1[8 + i] = orow[32 + c + 16 + i];
    }
  }

  v8f c4[4] = {};
#pragma unroll
  for (int kstep = 0; kstep < 2; ++kstep) {
    const float* wp = Wo + (size_t)(kstep * 32 + hi * 8) * N + nBase + ln;
#pragma unroll
    for (int nb = 0; nb < 4; ++nb) {
      v16bf bW;
#pragma unroll
      for (int i = 0; i < 8; ++i) { bW[i] = f2bf(wp[i * N + nb * 16]); bW[8 + i] = f2bf(wp[(16 + i) * N + nb * 16]); }
      c4[nb] = WMMA_BF16(kstep ? a1 : a0, bW, c4[nb]);
    }
  }

#pragma unroll
  for (int r = 0; r < 8; ++r) {
    float* yp = Y + (size_t)(rowBase + r + 8 * hi) * N + nBase;
#pragma unroll
    for (int nb = 0; nb < 4; ++nb)
      yp[nb * 16 + ln] = c4[nb][r] + bo[nBase + nb * 16 + ln];
  }
}

// ---------------------------------------------------------------------------
extern "C" void kernel_launch(void* const* d_in, const int* in_sizes, int n_in,
                              void* d_out, int out_size, void* d_ws, size_t ws_size,
                              hipStream_t stream) {
  constexpr int B = 8, S = 2048, DM = 1024, DK = 64;
  constexpr int M = B * S;

  const float* q    = (const float*)d_in[0];
  const float* k    = (const float*)d_in[1];
  const float* v    = (const float*)d_in[2];
  const int*   mask = (const int*)  d_in[3];
  const float* Wq   = (const float*)d_in[4];
  const float* bq   = (const float*)d_in[5];
  const float* Wk   = (const float*)d_in[6];
  const float* bk   = (const float*)d_in[7];
  const float* Wv   = (const float*)d_in[8];
  const float* bv   = (const float*)d_in[9];
  const float* Wo   = (const float*)d_in[10];
  const float* bo   = (const float*)d_in[11];

  const size_t projElems = (size_t)M * DK;
  __bf16* qpW = (__bf16*)d_ws;
  __bf16* kpW = qpW + projElems;
  __bf16* vpW = kpW + projElems;
  __bf16* oW  = vpW + projElems;

  dim3 blk(128);
  proj_gemm<<<dim3(M / 64), blk, 0, stream>>>(q, Wq, bq, qpW, DM);
  proj_gemm<<<dim3(M / 64), blk, 0, stream>>>(k, Wk, bk, kpW, DM);
  proj_gemm<<<dim3(M / 64), blk, 0, stream>>>(v, Wv, bv, vpW, DM);

  flash_attn<<<dim3(S / 64, B), blk, 0, stream>>>(qpW, kpW, vpW, mask, oW, S);

  out_proj<<<dim3(M / 16, DM / 256), blk, 0, stream>>>(oW, Wo, bo, (float*)d_out, DM);
}
